// LogicGatedSNN_22660247454003
// MI455X (gfx1250) — 8-bit matrix operand order repaired and checked by reading, compiled, not run on hardware
//
#include <hip/hip_runtime.h>

#define SNN_B   4096
#define SNN_IN  4096
#define SNN_OUT 4096
#define SNN_DECAY 0.8f
#define SNN_THRESH 1.0f

#define FP8_ONE     0x38
#define FP8_NEG_ONE 0xB8

typedef int   v16i __attribute__((ext_vector_type(16)));
typedef int   v4i  __attribute__((ext_vector_type(4)));
typedef int   v2i  __attribute__((ext_vector_type(2)));
typedef float v8f  __attribute__((ext_vector_type(8)));

__device__ __forceinline__ int fp8_spike(float v) {
  return v > 0.5f ? FP8_ONE : 0x00;
}
__device__ __forceinline__ int fp8_tern(float s) {
  return s > SNN_THRESH ? FP8_ONE : (s < -SNN_THRESH ? FP8_NEG_ONE : 0x00);
}

__global__ void snn_pack_spikes(const float* __restrict__ in, signed char* __restrict__ out) {
  size_t i = (size_t)blockIdx.x * blockDim.x + threadIdx.x;
  const float4* p = (const float4*)in + i * 4;
  int r[4];
#pragma unroll
  for (int j = 0; j < 4; ++j) {
    float4 v = p[j];
    r[j] = fp8_spike(v.x) | (fp8_spike(v.y) << 8) | (fp8_spike(v.z) << 16) | (fp8_spike(v.w) << 24);
  }
  v4i q = { r[0], r[1], r[2], r[3] };
  *((v4i*)out + i) = q;
}

__global__ void snn_pack_weights(const float* __restrict__ in, signed char* __restrict__ out) {
  size_t i = (size_t)blockIdx.x * blockDim.x + threadIdx.x;
  const float4* p = (const float4*)in + i * 4;
  int r[4];
#pragma unroll
  for (int j = 0; j < 4; ++j) {
    float4 v = p[j];
    r[j] = fp8_tern(v.x) | (fp8_tern(v.y) << 8) | (fp8_tern(v.z) << 16) | (fp8_tern(v.w) << 24);
  }
  v4i q = { r[0], r[1], r[2], r[3] };
  *((v4i*)out + i) = q;
}

__global__ void __launch_bounds__(256)
snn_wmma_kernel(const signed char* __restrict__ A8,
                const signed char* __restrict__ W8,
                const float* __restrict__ memb,
                const float* __restrict__ thr,
                const float* __restrict__ refr,
                float* __restrict__ out)
{
  const int K = SNN_IN;
  const int wave  = threadIdx.x >> 5;
  const int lane  = threadIdx.x & 31;
  const int wm    = wave & 3;
  const int wn    = wave >> 2;
  const int lhalf = lane >> 4;
  const int l16   = lane & 15;

  const int m0 = blockIdx.x * 128 + wm * 32;
  const int n0 = blockIdx.y * 64  + wn * 32;

  const signed char* a0 = A8 + (size_t)(m0 + l16) * K + lhalf * 8;
  const signed char* a1 = a0 + (size_t)16 * K;
  const signed char* b0 = W8 + (size_t)(n0 + l16) * K + lhalf * 8;
  const signed char* b1 = b0 + (size_t)16 * K;

  v8f acc00 = (v8f)(0.0f), acc01 = (v8f)(0.0f), acc10 = (v8f)(0.0f), acc11 = (v8f)(0.0f);

#pragma unroll 2
  for (int k = 0; k < K; k += 128) {
    v16i af0, af1, bf0, bf1;
    {
      v2i* d = (v2i*)&af0;
#pragma unroll
      for (int j = 0; j < 8; ++j) d[j] = *(const v2i*)(a0 + k + j * 16);
    }
    {
      v2i* d = (v2i*)&af1;
#pragma unroll
      for (int j = 0; j < 8; ++j) d[j] = *(const v2i*)(a1 + k + j * 16);
    }
    {
      v2i* q = (v2i*)&bf0;
#pragma unroll
      for (int j = 0; j < 8; ++j) q[j] = *(const v2i*)(b0 + k + j * 16);
    }
    {
      v2i* q = (v2i*)&bf1;
#pragma unroll
      for (int j = 0; j < 8; ++j) q[j] = *(const v2i*)(b1 + k + j * 16);
    }
    acc00 = __builtin_amdgcn_wmma_f32_16x16x128_fp8_fp8(af0, bf0, (short)0, acc00, false, false);
    acc01 = __builtin_amdgcn_wmma_f32_16x16x128_fp8_fp8(af0, bf1, (short)0, acc01, false, false);
    acc10 = __builtin_amdgcn_wmma_f32_16x16x128_fp8_fp8(af1, bf0, (short)0, acc10, false, false);
    acc11 = __builtin_amdgcn_wmma_f32_16x16x128_fp8_fp8(af1, bf1, (short)0, acc11, false, false);
  }

  const int mrow = lhalf * 8;
#pragma unroll
  for (int t = 0; t < 4; ++t) {
    const v8f& acc = (t == 0) ? acc00 : (t == 1) ? acc01 : (t == 2) ? acc10 : acc11;
    const int mt = m0 + ((t >> 1) ? 16 : 0);
    const int n  = n0 + ((t & 1) ? 16 : 0) + l16;
    const float gate = refr[n] > 0.0f ? 0.0f : 1.0f;
    const float base = memb[n] * SNN_DECAY;
    const float th   = thr[n];
#pragma unroll
    for (int r = 0; r < 8; ++r) {
      const int m = mt + mrow + r;
      const float v = base + acc[r] * gate;
      out[(size_t)m * SNN_OUT + n] = (v >= th) ? 1.0f : 0.0f;
    }
  }
}

extern "C" void kernel_launch(void* const* d_in, const int* in_sizes, int n_in,
                              void* d_out, int out_size, void* d_ws, size_t ws_size,
                              hipStream_t stream) {
  const float* spike = (const float*)d_in[0];
  const float* syn   = (const float*)d_in[1];
  const float* memb  = (const float*)d_in[2];
  const float* thr   = (const float*)d_in[3];
  const float* refr  = (const float*)d_in[4];
  float* out = (float*)d_out;

  signed char* A8 = (signed char*)d_ws;
  signed char* W8 = A8 + (size_t)SNN_B * SNN_IN;

  const int threads = 256;
  const int n16 = (SNN_B * SNN_IN) / 16;
  snn_pack_spikes<<<n16 / threads, threads, 0, stream>>>(spike, A8);
  snn_pack_weights<<<n16 / threads, threads, 0, stream>>>(syn, W8);

  dim3 grid(SNN_B / 128, SNN_OUT / 64);
  snn_wmma_kernel<<<grid, 256, 0, stream>>>(A8, W8, memb, thr, refr, out);
}
